// PSABlock_1494648619601
// MI455X (gfx1250) — matrix operand order repaired and checked by reading, not compiled, not run on hardware
//
#include <hip/hip_runtime.h>
#include <hip/hip_bf16.h>
#include <math.h>

typedef _Float16 f16;
typedef __attribute__((ext_vector_type(16))) _Float16 v16h;
typedef __attribute__((ext_vector_type(8)))  _Float16 v8h;
typedef __attribute__((ext_vector_type(8)))  float    v8f;
typedef __attribute__((ext_vector_type(4)))  float    v4f;

constexpr int cB  = 4;
constexpr int cC  = 256;
constexpr int cHH = 48;
constexpr int cWW = 48;
constexpr int cN  = cHH * cWW;   // 2304
constexpr int cNH = 4;
constexpr int cHD = 64;
constexpr int cKD = 32;
constexpr int cQKV = 512;
constexpr int cFF  = 512;

// ---------------------------------------------------------------- WMMA utils
__device__ inline v8f wmma16x16x32(v16h a, v16h b, v8f c) {
  return __builtin_amdgcn_wmma_f32_16x16x32_f16(false, a, false, b, (short)0, c, false, false);
}

__device__ inline v16h join8(v8h lo, v8h hi) {
  union { v16h w; v8h h[2]; } u; u.h[0] = lo; u.h[1] = hi; return u.w;
}

// A fragment 16x32 f16. Source row-major [row][k], ld elements.
// Layout: lane(0-15): M=lane, K = 0..15; lane(16-31): M=lane-16, K = 16..31
__device__ inline v16h load_afrag(const f16* base, int row, int ld, int k0, int lane) {
  int lh = lane & 15, half = lane >> 4;
  const f16* p = base + (size_t)(row + lh) * ld + k0 + 16 * half;
  return join8(*(const v8h*)p, *(const v8h*)(p + 8));
}

// B fragment 32x16 f16; "col" = WMMA-N index. Source row-major [col][k] (K-contiguous).
// lanes 0-15: K=0..15, lanes 16-31: K=16..31
__device__ inline v16h load_bfrag(const f16* base, int col, int ld, int k0, int lane) {
  int lh = lane & 15, half = lane >> 4;
  const f16* p = base + (size_t)(col + lh) * ld + k0 + 16 * half;
  return join8(*(const v8h*)p, *(const v8h*)(p + 8));
}

// B fragment from f32 source(s): convert (and optionally add two sources) on the fly.
template<bool DUAL>
__device__ inline v16h load_bfrag_f32(const float* b0, const float* b1, int col, int ld, int k0, int lane) {
  int lh = lane & 15, half = lane >> 4;
  size_t o = (size_t)(col + lh) * ld + k0 + 16 * half;
  v4f a0 = *(const v4f*)(b0 + o),     a1 = *(const v4f*)(b0 + o + 4),
      a2 = *(const v4f*)(b0 + o + 8), a3 = *(const v4f*)(b0 + o + 12);
  if (DUAL) {
    a0 += *(const v4f*)(b1 + o);     a1 += *(const v4f*)(b1 + o + 4);
    a2 += *(const v4f*)(b1 + o + 8); a3 += *(const v4f*)(b1 + o + 12);
  }
  v16h r;
#pragma unroll
  for (int i = 0; i < 4; i++) {
    r[i] = (f16)a0[i]; r[4 + i] = (f16)a1[i]; r[8 + i] = (f16)a2[i]; r[12 + i] = (f16)a3[i];
  }
  return r;
}

// Wave computes a 32(m) x 64(n) tile: acc[2][4], C layout m = 8*half + v.
template<int KTOT, int BMODE>
__device__ inline void gemm_core(const f16* A, int lda,
                                 const f16* B16, const float* Bf0, const float* Bf1, int ldb,
                                 int mbase, int nbase, int lane, v8f acc[2][4]) {
  for (int k0 = 0; k0 < KTOT; k0 += 32) {
    v16h a0 = load_afrag(A, mbase,      lda, k0, lane);
    v16h a1 = load_afrag(A, mbase + 16, lda, k0, lane);
#pragma unroll
    for (int f = 0; f < 4; f++) {
      v16h b;
      if constexpr (BMODE == 0)      b = load_bfrag(B16, nbase + f * 16, ldb, k0, lane);
      else if constexpr (BMODE == 1) b = load_bfrag_f32<false>(Bf0, nullptr, nbase + f * 16, ldb, k0, lane);
      else                           b = load_bfrag_f32<true>(Bf0, Bf1, nbase + f * 16, ldb, k0, lane);
      acc[0][f] = wmma16x16x32(a0, b, acc[0][f]);
      acc[1][f] = wmma16x16x32(a1, b, acc[1][f]);
    }
  }
}

// ---------------------------------------------------------------- pre-passes
__global__ void k_bnprep(const float* g, const float* b, const float* m, const float* v,
                         float* sc, float* bi, int n) {
  int i = blockIdx.x * 256 + threadIdx.x;
  if (i < n) {
    float s = g[i] * rsqrtf(v[i] + 1e-3f);
    sc[i] = s;
    bi[i] = b[i] - m[i] * s;
  }
}

__global__ void k_cvt16(const float* a, f16* o, int n) {
  int i = blockIdx.x * 256 + threadIdx.x;
  if (i < n) o[i] = (f16)a[i];
}

// x (b,c,n) f32 -> xT (b,n,c) in f32 and f16
__global__ void k_transpose(const float* __restrict__ x, float* __restrict__ xT32, f16* __restrict__ xT16) {
  __shared__ float tile[32][33];
  int b = blockIdx.z;
  int n0 = blockIdx.x * 32, c0 = blockIdx.y * 32;
  int tx = threadIdx.x, ty = threadIdx.y;
#pragma unroll
  for (int i = 0; i < 32; i += 8)
    tile[ty + i][tx] = x[((size_t)b * cC + c0 + ty + i) * cN + n0 + tx];
  __syncthreads();
#pragma unroll
  for (int i = 0; i < 32; i += 8) {
    float v = tile[tx][ty + i];
    size_t idx = ((size_t)b * cN + n0 + ty + i) * cC + c0 + tx;
    xT32[idx] = v;
    xT16[idx] = (f16)v;
  }
}

// ---------------------------------------------------------------- GEMM 1: QKV + BN, split epilogue
__global__ __launch_bounds__(256) void k_gemm_qkv(const f16* __restrict__ W, const f16* __restrict__ xT16,
                                                  const float* __restrict__ sc, const float* __restrict__ bi,
                                                  f16* __restrict__ qT, f16* __restrict__ kT,
                                                  f16* __restrict__ vT, float* __restrict__ v32) {
  int lane = threadIdx.x & 31, wave = threadIdx.x >> 5;
  int waveM = wave >> 1, waveN = wave & 1;
  int b = blockIdx.z;
  int mbase = blockIdx.x * 128 + waveM * 32;
  int nbase = blockIdx.y * 128 + waveN * 64;
  const f16* Bp = xT16 + (size_t)b * cN * cC;
  v8f acc[2][4] = {};
  gemm_core<256, 0>(W, cC, Bp, nullptr, nullptr, cC, mbase, nbase, lane, acc);
  int lh = lane & 15, half = lane >> 4;
#pragma unroll
  for (int i = 0; i < 2; i++)
#pragma unroll
    for (int f = 0; f < 4; f++)
#pragma unroll
      for (int v = 0; v < 8; v++) {
        int m = mbase + 16 * i + 8 * half + v;
        int n = nbase + 16 * f + lh;
        float val = acc[i][f][v] * sc[m] + bi[m];
        int h = m >> 7, j = m & 127;
        if (j < cKD) {
          // fold KD^-0.5 into q
          qT[(((size_t)b * cNH + h) * cN + n) * cKD + j] = (f16)(val * 0.17677669529663689f);
        } else if (j < 2 * cKD) {
          kT[(((size_t)b * cNH + h) * cN + n) * cKD + (j - cKD)] = (f16)val;
        } else {
          int d = j - 2 * cKD;
          vT[(((size_t)b * cNH + h) * cHD + d) * cN + n] = (f16)val;           // (b,h,d,m) m-contig: PV B-operand
          v32[((size_t)b * cC + h * cHD + d) * cN + n] = val;                  // spatial layout for dwconv
        }
      }
}

// ---------------------------------------------------------------- depthwise 3x3 + BN -> peT (b,n,c) f32
__global__ void k_pe(const float* __restrict__ v32, const float* __restrict__ pw,
                     const float* __restrict__ sc, const float* __restrict__ bi,
                     float* __restrict__ peT) {
  int c = threadIdx.x;
  int n = blockIdx.x;
  int b = blockIdx.y;
  int y = n / cWW, x = n % cWW;
  const float* vc = v32 + ((size_t)b * cC + c) * cN;
  const float* wc = pw + c * 9;
  float acc = 0.f;
#pragma unroll
  for (int dy = 0; dy < 3; dy++) {
    int yy = y + dy - 1;
    if (yy < 0 || yy >= cHH) continue;
#pragma unroll
    for (int dx = 0; dx < 3; dx++) {
      int xx = x + dx - 1;
      if (xx < 0 || xx >= cWW) continue;
      acc += vc[yy * cWW + xx] * wc[dy * 3 + dx];
    }
  }
  peT[((size_t)b * cN + n) * cC + c] = acc * sc[c] + bi[c];
}

// ---------------------------------------------------------------- Flash attention
// grid (N/128, NH, B), 256 threads = 8 waves, each wave owns 16 query rows.
__global__ __launch_bounds__(256) void k_attn(const f16* __restrict__ qT, const f16* __restrict__ kTg,
                                              const f16* __restrict__ vT, float* __restrict__ oT) {
  __shared__ __align__(16) f16 kTile[128][40];       // [m][kd]
  __shared__ __align__(16) f16 vTile[64][136];       // [d][m]
  __shared__ __align__(16) f16 pBuf[8][16][136];     // per-wave P strip [row][m]

  int lane = threadIdx.x & 31, wave = threadIdx.x >> 5;
  int lh = lane & 15, half = lane >> 4;
  int nblock = blockIdx.x * 128;
  int h = blockIdx.y, b = blockIdx.z;

  const f16* q  = qT  + (((size_t)b * cNH + h) * cN) * cKD;
  const f16* kk = kTg + (((size_t)b * cNH + h) * cN) * cKD;
  const f16* vp = vT  + (((size_t)b * cNH + h) * cHD) * cN;

  // Q fragment (16 rows x 32 kd) stays in registers all kernel.
  v16h qf = load_afrag(q, nblock + wave * 16, cKD, 0, lane);

  v8f o[4] = {};
  float rowm[8], rowl[8];
#pragma unroll
  for (int v = 0; v < 8; v++) { rowm[v] = -1e30f; rowl[v] = 0.f; }

  for (int m0 = 0; m0 < cN; m0 += 128) {
    __syncthreads();
    {
      int t = threadIdx.x;
      // K tile: 128 rows x 32 f16 (64B/row), 32B per thread
      int row = t >> 1, off = (t & 1) * 16;
      *(v8h*)&kTile[row][off]     = *(const v8h*)(kk + (size_t)(m0 + row) * cKD + off);
      *(v8h*)&kTile[row][off + 8] = *(const v8h*)(kk + (size_t)(m0 + row) * cKD + off + 8);
      // V tile: 64 rows x 128 f16 (256B/row), 64B per thread
      int vr = t >> 2, vo = (t & 3) * 32;
#pragma unroll
      for (int u = 0; u < 4; u++)
        *(v8h*)&vTile[vr][vo + u * 8] = *(const v8h*)(vp + (size_t)vr * cN + m0 + vo + u * 8);
    }
    __syncthreads();

    // S = Q K^T  (8 tiles of 16x16, K-dim = 32 = one WMMA)
    v8f s[8];
#pragma unroll
    for (int f = 0; f < 8; f++) {
      v8f z = {0.f, 0.f, 0.f, 0.f, 0.f, 0.f, 0.f, 0.f};
      v16h bf = load_bfrag(&kTile[0][0], f * 16, 40, 0, lane);
      s[f] = wmma16x16x32(qf, bf, z);
    }

    // online softmax: row r = 8*half + v lives in 16 lanes of one half
    float tm[8];
#pragma unroll
    for (int v = 0; v < 8; v++) {
      float mx = s[0][v];
#pragma unroll
      for (int f = 1; f < 8; f++) mx = fmaxf(mx, s[f][v]);
#pragma unroll
      for (int st = 1; st < 16; st <<= 1) mx = fmaxf(mx, __shfl_xor(mx, st, 32));
      tm[v] = mx;
    }
    float alpha[8];
#pragma unroll
    for (int v = 0; v < 8; v++) {
      float mn = fmaxf(rowm[v], tm[v]);
      alpha[v] = __expf(rowm[v] - mn);
      rowm[v] = mn;
    }
    float rs[8];
#pragma unroll
    for (int v = 0; v < 8; v++) rs[v] = 0.f;
#pragma unroll
    for (int f = 0; f < 8; f++)
#pragma unroll
      for (int v = 0; v < 8; v++) {
        float p = __expf(s[f][v] - rowm[v]);
        s[f][v] = p;
        rs[v] += p;
      }
#pragma unroll
    for (int v = 0; v < 8; v++) {
#pragma unroll
      for (int st = 1; st < 16; st <<= 1) rs[v] += __shfl_xor(rs[v], st, 32);
      rowl[v] = rowl[v] * alpha[v] + rs[v];
    }
#pragma unroll
    for (int dt = 0; dt < 4; dt++)
#pragma unroll
      for (int v = 0; v < 8; v++) o[dt][v] *= alpha[v];

    // P: C-layout -> A-layout via per-wave LDS strip (no block barrier needed)
#pragma unroll
    for (int f = 0; f < 8; f++)
#pragma unroll
      for (int v = 0; v < 8; v++)
        pBuf[wave][8 * half + v][f * 16 + lh] = (f16)s[f][v];

    // O += P (16x128) x V^T (128x64): K-dim 128 = 4 WMMA steps
    v16h pa[4];
#pragma unroll
    for (int kt = 0; kt < 4; kt++)
      pa[kt] = load_afrag(&pBuf[wave][0][0], 0, 136, kt * 32, lane);
#pragma unroll
    for (int dt = 0; dt < 4; dt++)
#pragma unroll
      for (int kt = 0; kt < 4; kt++) {
        v16h bv = load_bfrag(&vTile[0][0], dt * 16, 136, kt * 32, lane);
        o[dt] = wmma16x16x32(pa[kt], bv, o[dt]);
      }
  }

  // normalize + write oT (b, n, c) f32  (B-operand layout for proj GEMM)
  float inv[8];
#pragma unroll
  for (int v = 0; v < 8; v++) inv[v] = 1.f / rowl[v];
#pragma unroll
  for (int dt = 0; dt < 4; dt++)
#pragma unroll
    for (int v = 0; v < 8; v++) {
      int n = nblock + wave * 16 + 8 * half + v;
      int c = h * cHD + dt * 16 + lh;
      oT[((size_t)b * cN + n) * cC + c] = o[dt][v] * inv[v];
    }
}

// ---------------------------------------------------------------- GEMM 2: proj((o+pe)) + BN + residual
__global__ __launch_bounds__(256) void k_gemm_proj(const f16* __restrict__ W,
                                                   const float* __restrict__ oT, const float* __restrict__ peT,
                                                   const float* __restrict__ xT32,
                                                   const float* __restrict__ sc, const float* __restrict__ bi,
                                                   float* __restrict__ x1T32, f16* __restrict__ x1T16) {
  int lane = threadIdx.x & 31, wave = threadIdx.x >> 5;
  int waveM = wave >> 1, waveN = wave & 1;
  int b = blockIdx.z;
  int mbase = blockIdx.x * 128 + waveM * 32;
  int nbase = blockIdx.y * 128 + waveN * 64;
  const float* B0 = oT  + (size_t)b * cN * cC;
  const float* B1 = peT + (size_t)b * cN * cC;
  v8f acc[2][4] = {};
  gemm_core<256, 2>(W, cC, nullptr, B0, B1, cC, mbase, nbase, lane, acc);
  int lh = lane & 15, half = lane >> 4;
#pragma unroll
  for (int i = 0; i < 2; i++)
#pragma unroll
    for (int f = 0; f < 4; f++)
#pragma unroll
      for (int v = 0; v < 8; v++) {
        int m = mbase + 16 * i + 8 * half + v;
        int n = nbase + 16 * f + lh;
        float val = acc[i][f][v] * sc[m] + bi[m];
        size_t idx = ((size_t)b * cN + n) * cC + m;
        float x1 = xT32[idx] + val;
        x1T32[idx] = x1;
        x1T16[idx] = (f16)x1;
      }
}

// ---------------------------------------------------------------- GEMM 3: f1 + BN + SiLU
__global__ __launch_bounds__(256) void k_gemm_f1(const f16* __restrict__ W, const f16* __restrict__ x1T16,
                                                 const float* __restrict__ sc, const float* __restrict__ bi,
                                                 f16* __restrict__ fT) {
  int lane = threadIdx.x & 31, wave = threadIdx.x >> 5;
  int waveM = wave >> 1, waveN = wave & 1;
  int b = blockIdx.z;
  int mbase = blockIdx.x * 128 + waveM * 32;
  int nbase = blockIdx.y * 128 + waveN * 64;
  const f16* Bp = x1T16 + (size_t)b * cN * cC;
  v8f acc[2][4] = {};
  gemm_core<256, 0>(W, cC, Bp, nullptr, nullptr, cC, mbase, nbase, lane, acc);
  int lh = lane & 15, half = lane >> 4;
#pragma unroll
  for (int i = 0; i < 2; i++)
#pragma unroll
    for (int f = 0; f < 4; f++)
#pragma unroll
      for (int v = 0; v < 8; v++) {
        int m = mbase + 16 * i + 8 * half + v;
        int n = nbase + 16 * f + lh;
        float val = acc[i][f][v] * sc[m] + bi[m];
        val = val / (1.f + __expf(-val));   // SiLU
        fT[((size_t)b * cN + n) * cFF + m] = (f16)val;
      }
}

// ---------------------------------------------------------------- GEMM 4: f2 + BN + residual -> out (b,c,h,w)
__global__ __launch_bounds__(256) void k_gemm_f2(const f16* __restrict__ W, const f16* __restrict__ fT,
                                                 const float* __restrict__ x1T32,
                                                 const float* __restrict__ sc, const float* __restrict__ bi,
                                                 float* __restrict__ out) {
  int lane = threadIdx.x & 31, wave = threadIdx.x >> 5;
  int waveM = wave >> 1, waveN = wave & 1;
  int b = blockIdx.z;
  int mbase = blockIdx.x * 128 + waveM * 32;
  int nbase = blockIdx.y * 128 + waveN * 64;
  const f16* Bp = fT + (size_t)b * cN * cFF;
  v8f acc[2][4] = {};
  gemm_core<512, 0>(W, cFF, Bp, nullptr, nullptr, cFF, mbase, nbase, lane, acc);
  int lh = lane & 15, half = lane >> 4;
#pragma unroll
  for (int i = 0; i < 2; i++)
#pragma unroll
    for (int f = 0; f < 4; f++)
#pragma unroll
      for (int v = 0; v < 8; v++) {
        int m = mbase + 16 * i + 8 * half + v;
        int n = nbase + 16 * f + lh;
        float val = acc[i][f][v] * sc[m] + bi[m];
        out[((size_t)b * cC + m) * cN + n] = x1T32[((size_t)b * cN + n) * cC + m] + val;
      }
}

// ---------------------------------------------------------------- launcher
extern "C" void kernel_launch(void* const* d_in, const int* in_sizes, int n_in,
                              void* d_out, int out_size, void* d_ws, size_t ws_size,
                              hipStream_t stream) {
  (void)in_sizes; (void)n_in; (void)out_size; (void)ws_size;
  const float* x      = (const float*)d_in[0];
  const float* qkv_w  = (const float*)d_in[1];
  const float* qkv_g  = (const float*)d_in[2];
  const float* qkv_b  = (const float*)d_in[3];
  const float* qkv_m  = (const float*)d_in[4];
  const float* qkv_v  = (const float*)d_in[5];
  const float* pe_w   = (const float*)d_in[6];
  const float* pe_g   = (const float*)d_in[7];
  const float* pe_b   = (const float*)d_in[8];
  const float* pe_m   = (const float*)d_in[9];
  const float* pe_v   = (const float*)d_in[10];
  const float* proj_w = (const float*)d_in[11];
  const float* proj_g = (const float*)d_in[12];
  const float* proj_b = (const float*)d_in[13];
  const float* proj_m = (const float*)d_in[14];
  const float* proj_v = (const float*)d_in[15];
  const float* f1_w   = (const float*)d_in[16];
  const float* f1_g   = (const float*)d_in[17];
  const float* f1_b   = (const float*)d_in[18];
  const float* f1_m   = (const float*)d_in[19];
  const float* f1_v   = (const float*)d_in[20];
  const float* f2_w   = (const float*)d_in[21];
  const float* f2_g   = (const float*)d_in[22];
  const float* f2_b   = (const float*)d_in[23];
  const float* f2_m   = (const float*)d_in[24];
  const float* f2_v   = (const float*)d_in[25];
  float* out = (float*)d_out;

  char* base = (char*)d_ws;
  size_t off = 0;
  auto alloc = [&](size_t bytes) -> void* {
    void* p = base + off;
    off += (bytes + 255) & ~(size_t)255;
    return p;
  };

  const size_t nBNC = (size_t)cB * cN * cC;
  float* xT32  = (float*)alloc(nBNC * 4);
  f16*   xT16  = (f16*)  alloc(nBNC * 2);
  f16*   qT    = (f16*)  alloc((size_t)cB * cNH * cN * cKD * 2);
  f16*   kT    = (f16*)  alloc((size_t)cB * cNH * cN * cKD * 2);
  f16*   vT    = (f16*)  alloc(nBNC * 2);
  float* v32   = (float*)alloc(nBNC * 4);
  float* peT   = (float*)alloc(nBNC * 4);
  float* oT    = (float*)alloc(nBNC * 4);
  float* x1T32 = (float*)alloc(nBNC * 4);
  f16*   x1T16 = (f16*)  alloc(nBNC * 2);
  f16*   fT    = (f16*)  alloc((size_t)cB * cN * cFF * 2);
  f16*   wq16  = (f16*)  alloc((size_t)cQKV * cC * 2);
  f16*   wp16  = (f16*)  alloc((size_t)cC * cC * 2);
  f16*   w116  = (f16*)  alloc((size_t)cFF * cC * 2);
  f16*   w216  = (f16*)  alloc((size_t)cC * cFF * 2);
  float* sc_qkv  = (float*)alloc(cQKV * 4); float* bi_qkv  = (float*)alloc(cQKV * 4);
  float* sc_pe   = (float*)alloc(cC * 4);   float* bi_pe   = (float*)alloc(cC * 4);
  float* sc_proj = (float*)alloc(cC * 4);   float* bi_proj = (float*)alloc(cC * 4);
  float* sc_f1   = (float*)alloc(cFF * 4);  float* bi_f1   = (float*)alloc(cFF * 4);
  float* sc_f2   = (float*)alloc(cC * 4);   float* bi_f2   = (float*)alloc(cC * 4);

  // BN folds
  k_bnprep<<<2, 256, 0, stream>>>(qkv_g, qkv_b, qkv_m, qkv_v, sc_qkv, bi_qkv, cQKV);
  k_bnprep<<<1, 256, 0, stream>>>(pe_g, pe_b, pe_m, pe_v, sc_pe, bi_pe, cC);
  k_bnprep<<<1, 256, 0, stream>>>(proj_g, proj_b, proj_m, proj_v, sc_proj, bi_proj, cC);
  k_bnprep<<<2, 256, 0, stream>>>(f1_g, f1_b, f1_m, f1_v, sc_f1, bi_f1, cFF);
  k_bnprep<<<1, 256, 0, stream>>>(f2_g, f2_b, f2_m, f2_v, sc_f2, bi_f2, cC);

  // weight f32 -> f16
  k_cvt16<<<(cQKV * cC + 255) / 256, 256, 0, stream>>>(qkv_w, wq16, cQKV * cC);
  k_cvt16<<<(cC * cC + 255) / 256, 256, 0, stream>>>(proj_w, wp16, cC * cC);
  k_cvt16<<<(cFF * cC + 255) / 256, 256, 0, stream>>>(f1_w, w116, cFF * cC);
  k_cvt16<<<(cC * cFF + 255) / 256, 256, 0, stream>>>(f2_w, w216, cC * cFF);

  // x -> token-major
  k_transpose<<<dim3(cN / 32, cC / 32, cB), dim3(32, 8), 0, stream>>>(x, xT32, xT16);

  // QKV GEMM (M=512)
  k_gemm_qkv<<<dim3(cQKV / 128, cN / 128, cB), 256, 0, stream>>>(wq16, xT16, sc_qkv, bi_qkv, qT, kT, vT, v32);

  // depthwise positional encoding
  k_pe<<<dim3(cN, cB), cC, 0, stream>>>(v32, pe_w, sc_pe, bi_pe, peT);

  // flash attention
  k_attn<<<dim3(cN / 128, cNH, cB), 256, 0, stream>>>(qT, kT, vT, oT);

  // proj GEMM + residual (M=256)
  k_gemm_proj<<<dim3(cC / 128, cN / 128, cB), 256, 0, stream>>>(wp16, oT, peT, xT32, sc_proj, bi_proj, x1T32, x1T16);

  // FFN
  k_gemm_f1<<<dim3(cFF / 128, cN / 128, cB), 256, 0, stream>>>(w116, x1T16, sc_f1, bi_f1, fT);
  k_gemm_f2<<<dim3(cC / 128, cN / 128, cB), 256, 0, stream>>>(w216, fT, x1T32, sc_f2, bi_f2, out);
}
